// NextWordLSTM_30322469110224
// MI455X (gfx1250) — compile-verified, matrix operand order checked by reading, not run on hardware
//
#include <hip/hip_runtime.h>
#include <hip/hip_bf16.h>

typedef __attribute__((ext_vector_type(16))) __bf16 v16bf;
typedef __attribute__((ext_vector_type(8)))  __bf16 v8bf;
typedef __attribute__((ext_vector_type(8)))  float  v8f;

#define BATCH  64
#define SEQ    128
#define EMBED  512
#define HIDDEN 1024
#define GATES  4096
#define VOCAB  32000

// ---------------------------------------------------------------------------
// WMMA helpers
// ---------------------------------------------------------------------------
// 16x32 bf16 fragment (ISA 7.12.2): lane L holds row (L&15); lanes 0-15 carry
// K = +0..7 and +16..23, lanes 16-31 carry K = +8..15 and +24..31.
// p must already point at element (row0 + (lane&15))*ld + k0 + ((lane>>4)<<3).
__device__ __forceinline__ v16bf load_frag(const __bf16* __restrict__ p) {
  v8bf lo = *(const v8bf*)(p);        // 16B
  v8bf hi = *(const v8bf*)(p + 16);   // 16B, K+16 group
  return __builtin_shufflevector(lo, hi, 0,1,2,3,4,5,6,7,8,9,10,11,12,13,14,15);
}

__device__ __forceinline__ v8f wmma_bf16(v16bf a, v16bf b, v8f c) {
  return __builtin_amdgcn_wmma_f32_16x16x32_bf16(false, a, false, b,
                                                 (short)0, c, false, false);
}

__device__ __forceinline__ float sigmoidf_(float x) {
  return 1.0f / (1.0f + __expf(-x));
}

// ---------------------------------------------------------------------------
// Utility kernels
// ---------------------------------------------------------------------------
__global__ void zero_u32_kernel(unsigned* p, long long n) {
  long long i = (long long)blockIdx.x * blockDim.x + threadIdx.x;
  if (i < n) p[i] = 0u;
}

__global__ void cvt_bf16_kernel(const float* __restrict__ s,
                                __bf16* __restrict__ d, long long n) {
  long long i = (long long)blockIdx.x * blockDim.x + threadIdx.x;
  if (i < n) d[i] = (__bf16)s[i];
}

// One block per (b,t) row: gather embedding row and narrow to bf16.
__global__ void embed_kernel(const int* __restrict__ x,
                             const float* __restrict__ emb,
                             __bf16* __restrict__ xe) {
  int row = blockIdx.x;                       // b*SEQ + t
  int tok = x[row];
  const float4* src = (const float4*)(emb + (size_t)tok * EMBED);
  float4 v = src[threadIdx.x];                // 128 threads * 4 = 512
  __bf16* d = xe + (size_t)row * EMBED + threadIdx.x * 4;
  d[0] = (__bf16)v.x; d[1] = (__bf16)v.y; d[2] = (__bf16)v.z; d[3] = (__bf16)v.w;
}

__global__ void count_kernel(const int* __restrict__ x,
                             unsigned* __restrict__ counts) {
  int i = blockIdx.x * blockDim.x + threadIdx.x;
  if (i < BATCH * SEQ) {
    int b = i / SEQ;
    atomicAdd(&counts[(size_t)b * VOCAB + x[i]], 1u);
  }
}

// ---------------------------------------------------------------------------
// Generic GEMM: C[M,N] = A[M,K](bf16) * W[N,K]^T(bf16) + b0[n] + b1[n]
// Wave tile: 64(M) x 64(N) -> 16 WMMAs per 8 fragment loads per K-step.
// K is a template constant so all tile displacements fold into the 24-bit
// immediate offset field; only two 32-bit offsets advance per iteration.
// ---------------------------------------------------------------------------
template<int K>
__global__ __launch_bounds__(128)
void gemm_bias_kernel(const __bf16* __restrict__ A, const __bf16* __restrict__ W,
                      const float* __restrict__ b0, const float* __restrict__ b1,
                      float* __restrict__ C, int N, int tiles_m) {
  int lane = threadIdx.x & 31;
  int wid  = blockIdx.x * 4 + (threadIdx.x >> 5);
  int row0 = (wid % tiles_m) * 64;
  int col0 = (wid / tiles_m) * 64;

  unsigned lr = lane & 15, lk = (unsigned)(lane >> 4) << 3;
  unsigned aoff = (unsigned)(row0 + lr) * K + lk;   // element offsets
  unsigned boff = (unsigned)(col0 + lr) * K + lk;

  v8f acc[4][4] = {};
#pragma unroll 1
  for (int k = 0; k < K; k += 32, aoff += 32, boff += 32) {
    v16bf af[4], bf[4];
#pragma unroll
    for (int i = 0; i < 4; ++i) af[i] = load_frag(A + aoff + i * 16 * K);
#pragma unroll
    for (int j = 0; j < 4; ++j) bf[j] = load_frag(W + boff + j * 16 * K);
#pragma unroll
    for (int i = 0; i < 4; ++i)
#pragma unroll
      for (int j = 0; j < 4; ++j)
        acc[i][j] = wmma_bf16(af[i], bf[j], acc[i][j]);
  }

  unsigned cr = (unsigned)(lane >> 4) * 8, cc = lane & 15;
#pragma unroll
  for (int i = 0; i < 4; ++i)
#pragma unroll
    for (int j = 0; j < 4; ++j) {
      unsigned col = col0 + j * 16 + cc;
      float bias = b0[col] + b1[col];
#pragma unroll
      for (int r = 0; r < 8; ++r) {
        unsigned row = row0 + i * 16 + cr + r;
        C[(size_t)row * N + col] = acc[i][j][r] + bias;
      }
    }
}

// ---------------------------------------------------------------------------
// One LSTM timestep with 2-way split-K:
//   gates = xproj[:,t,:] + h_in @ whh^T ; elementwise LSTM update.
// Grid: 64 blocks (16 hidden cols) x 8 waves. Waves 0-3 (kh=0) cover batch
// tiles for K=0..511; waves 4-7 (kh=1) cover the same tiles for K=512..1023
// and push their partial accumulators through LDS. Waves 0-3 reduce and run
// the activation epilogue in-register off the WMMA C-layout (i,f,g,o for the
// same (b,j) sit at the same accumulator slot).
// ---------------------------------------------------------------------------
__global__ __launch_bounds__(256)
void lstm_step_kernel(const float* __restrict__ xproj,
                      const __bf16* __restrict__ whh,
                      const __bf16* __restrict__ h_in,
                      __bf16* __restrict__ h_out,
                      float* __restrict__ c,
                      __bf16* __restrict__ hs_out, int t) {
  __shared__ v8f red[4][4][32];         // [m_tile][gate][lane] = 16 KB

  int lane = threadIdx.x & 31;
  int wave = threadIdx.x >> 5;          // 0..7
  int mt   = wave & 3;                  // batch tile index
  int kh   = wave >> 2;                 // K half
  int m0   = mt * 16;
  int j0   = blockIdx.x * 16;           // hidden tile

  unsigned lr = lane & 15, lk = (unsigned)(lane >> 4) << 3;
  unsigned aoff = (unsigned)(m0 + lr) * HIDDEN + lk + kh * (HIDDEN / 2);
  unsigned boff = (unsigned)(j0 + lr) * HIDDEN + lk + kh * (HIDDEN / 2);

  v8f acc[4] = {};
#pragma unroll 2
  for (int it = 0; it < HIDDEN / 64; ++it, aoff += 32, boff += 32) {
    v16bf a = load_frag(h_in + aoff);
    v16bf bf[4];
#pragma unroll
    for (int g = 0; g < 4; ++g)
      bf[g] = load_frag(whh + boff + g * (HIDDEN * HIDDEN));
#pragma unroll
    for (int g = 0; g < 4; ++g) acc[g] = wmma_bf16(a, bf[g], acc[g]);
  }

  if (kh == 1) {
#pragma unroll
    for (int g = 0; g < 4; ++g) red[mt][g][lane] = acc[g];
  }
  __syncthreads();
  if (kh == 1) return;

#pragma unroll
  for (int g = 0; g < 4; ++g) acc[g] += red[mt][g][lane];

  unsigned cr = (unsigned)(lane >> 4) * 8;
  unsigned j  = j0 + (lane & 15);
#pragma unroll
  for (int r = 0; r < 8; ++r) {
    unsigned b = m0 + cr + r;
    size_t xb = ((size_t)b * SEQ + t) * GATES + j;
    float ig = sigmoidf_(acc[0][r] + xproj[xb]);
    float fg = sigmoidf_(acc[1][r] + xproj[xb + HIDDEN]);
    float gg = tanhf    (acc[2][r] + xproj[xb + 2 * HIDDEN]);
    float og = sigmoidf_(acc[3][r] + xproj[xb + 3 * HIDDEN]);
    size_t ci = (size_t)b * HIDDEN + j;
    float cn = fg * c[ci] + ig * gg;
    float hn = og * tanhf(cn);
    c[ci] = cn;
    h_out[ci] = (__bf16)hn;
    if (hs_out) hs_out[((size_t)b * SEQ + t) * HIDDEN + j] = (__bf16)hn;
  }
}

// ---------------------------------------------------------------------------
// FC + repetition penalty: out = (h @ fc_w^T + fc_b) * 1.2^(-count)
// M = 64 exactly -> single 64-row band, wave tile 64x64.
// ---------------------------------------------------------------------------
__global__ __launch_bounds__(128)
void fc_penalty_kernel(const __bf16* __restrict__ A, const __bf16* __restrict__ W,
                       const float* __restrict__ bias,
                       const unsigned* __restrict__ counts,
                       float* __restrict__ C) {
  const int K = HIDDEN, N = VOCAB;
  (void)N;
  int lane = threadIdx.x & 31;
  int wid  = blockIdx.x * 4 + (threadIdx.x >> 5);
  int col0 = wid * 64;                  // tiles_m == 1, row0 == 0

  unsigned lr = lane & 15, lk = (unsigned)(lane >> 4) << 3;
  unsigned aoff = lr * K + lk;
  unsigned boff = (unsigned)(col0 + lr) * K + lk;

  v8f acc[4][4] = {};
#pragma unroll 1
  for (int k = 0; k < K; k += 32, aoff += 32, boff += 32) {
    v16bf af[4], bf[4];
#pragma unroll
    for (int i = 0; i < 4; ++i) af[i] = load_frag(A + aoff + i * 16 * K);
#pragma unroll
    for (int j = 0; j < 4; ++j) bf[j] = load_frag(W + boff + j * 16 * K);
#pragma unroll
    for (int i = 0; i < 4; ++i)
#pragma unroll
      for (int j = 0; j < 4; ++j)
        acc[i][j] = wmma_bf16(af[i], bf[j], acc[i][j]);
  }

  const float nlog2_pen = -0.26303440583379378f; // -log2(1.2)
  unsigned cr = (unsigned)(lane >> 4) * 8, cc = lane & 15;
#pragma unroll
  for (int i = 0; i < 4; ++i)
#pragma unroll
    for (int j = 0; j < 4; ++j) {
      unsigned col = col0 + j * 16 + cc;
      float bv = bias[col];
#pragma unroll
      for (int r = 0; r < 8; ++r) {
        unsigned row = i * 16 + cr + r;
        unsigned cnt = counts[(size_t)row * VOCAB + col];
        float scale = exp2f((float)cnt * nlog2_pen);
        C[(size_t)row * VOCAB + col] = (acc[i][j][r] + bv) * scale;
      }
    }
}

// ---------------------------------------------------------------------------
// Host driver
// ---------------------------------------------------------------------------
static inline void launch_zero(unsigned* p, long long n, hipStream_t s) {
  zero_u32_kernel<<<(unsigned)((n + 255) / 256), 256, 0, s>>>(p, n);
}
static inline void launch_cvt(const float* src, __bf16* dst, long long n, hipStream_t s) {
  cvt_bf16_kernel<<<(unsigned)((n + 255) / 256), 256, 0, s>>>(src, dst, n);
}

extern "C" void kernel_launch(void* const* d_in, const int* in_sizes, int n_in,
                              void* d_out, int out_size, void* d_ws, size_t ws_size,
                              hipStream_t stream) {
  const int*   x    = (const int*)  d_in[0];
  const float* emb  = (const float*)d_in[1];
  const float* wih0 = (const float*)d_in[2];
  const float* whh0 = (const float*)d_in[3];
  const float* bih0 = (const float*)d_in[4];
  const float* bhh0 = (const float*)d_in[5];
  const float* wih1 = (const float*)d_in[6];
  const float* whh1 = (const float*)d_in[7];
  const float* bih1 = (const float*)d_in[8];
  const float* bhh1 = (const float*)d_in[9];
  const float* fcw  = (const float*)d_in[10];
  const float* fcb  = (const float*)d_in[11];
  float* out = (float*)d_out;

  char* ws = (char*)d_ws;
  size_t off = 0;
  auto alloc = [&](size_t bytes) -> char* {
    char* p = ws + off; off += (bytes + 255) & ~(size_t)255; return p;
  };
  __bf16*   xemb   = (__bf16*)alloc((size_t)BATCH * SEQ * EMBED * 2);
  float*    xproj  = (float*) alloc((size_t)BATCH * SEQ * GATES * 4);  // reused L0/L1
  __bf16*   hs0    = (__bf16*)alloc((size_t)BATCH * SEQ * HIDDEN * 2);
  __bf16*   wih0b  = (__bf16*)alloc((size_t)GATES * EMBED * 2);
  __bf16*   whh0b  = (__bf16*)alloc((size_t)GATES * HIDDEN * 2);
  __bf16*   wih1b  = (__bf16*)alloc((size_t)GATES * HIDDEN * 2);
  __bf16*   whh1b  = (__bf16*)alloc((size_t)GATES * HIDDEN * 2);
  __bf16*   fcwb   = (__bf16*)alloc((size_t)VOCAB * HIDDEN * 2);
  __bf16*   hA     = (__bf16*)alloc((size_t)BATCH * HIDDEN * 2);
  __bf16*   hB     = (__bf16*)alloc((size_t)BATCH * HIDDEN * 2);
  float*    cbuf   = (float*) alloc((size_t)BATCH * HIDDEN * 4);
  unsigned* counts = (unsigned*)alloc((size_t)BATCH * VOCAB * 4);

  // 1) weights -> bf16
  launch_cvt(wih0, wih0b, (long long)GATES * EMBED,  stream);
  launch_cvt(whh0, whh0b, (long long)GATES * HIDDEN, stream);
  launch_cvt(wih1, wih1b, (long long)GATES * HIDDEN, stream);
  launch_cvt(whh1, whh1b, (long long)GATES * HIDDEN, stream);
  launch_cvt(fcw,  fcwb,  (long long)VOCAB * HIDDEN, stream);

  // 2) embedding gather
  embed_kernel<<<BATCH * SEQ, 128, 0, stream>>>(x, emb, xemb);

  // 3) layer-0 x_proj: [8192,4096] = xemb[8192,512] @ wih0^T + b_ih0 + b_hh0
  {
    int tiles_m = (BATCH * SEQ) / 64, tiles_n = GATES / 64;   // 128 x 64
    gemm_bias_kernel<EMBED><<<(tiles_m * tiles_n) / 4, 128, 0, stream>>>(
        xemb, wih0b, bih0, bhh0, xproj, GATES, tiles_m);
  }

  // 4) layer-0 recurrence
  launch_zero((unsigned*)hA,   (long long)BATCH * HIDDEN / 2, stream); // bf16
  launch_zero((unsigned*)cbuf, (long long)BATCH * HIDDEN,     stream);
  {
    __bf16 *hin = hA, *hout = hB;
    for (int t = 0; t < SEQ; ++t) {
      lstm_step_kernel<<<HIDDEN / 16, 256, 0, stream>>>(
          xproj, whh0b, hin, hout, cbuf, hs0, t);
      __bf16* tmp = hin; hin = hout; hout = tmp;
    }
  }

  // 5) layer-1 x_proj: hs0[8192,1024] @ wih1^T + b_ih1 + b_hh1 (reuse xproj)
  {
    int tiles_m = (BATCH * SEQ) / 64, tiles_n = GATES / 64;
    gemm_bias_kernel<HIDDEN><<<(tiles_m * tiles_n) / 4, 128, 0, stream>>>(
        hs0, wih1b, bih1, bhh1, xproj, GATES, tiles_m);
  }

  // 6) layer-1 recurrence (final h ends up in hA after 128 swaps)
  launch_zero((unsigned*)hA,   (long long)BATCH * HIDDEN / 2, stream);
  launch_zero((unsigned*)cbuf, (long long)BATCH * HIDDEN,     stream);
  {
    __bf16 *hin = hA, *hout = hB;
    for (int t = 0; t < SEQ; ++t) {
      lstm_step_kernel<<<HIDDEN / 16, 256, 0, stream>>>(
          xproj, whh1b, hin, hout, cbuf, (__bf16*)nullptr, t);
      __bf16* tmp = hin; hin = hout; hout = tmp;
    }
  }

  // 7) repetition counts
  launch_zero(counts, (long long)BATCH * VOCAB, stream);
  count_kernel<<<(BATCH * SEQ + 255) / 256, 256, 0, stream>>>(x, counts);

  // 8) FC + penalty -> d_out [64, 32000]
  {
    int waves = VOCAB / 64;                  // 500
    fc_penalty_kernel<<<waves / 4, 128, 0, stream>>>(hA, fcwb, fcb, counts, out);
  }
}
